// DAWNLayer_87677462380611
// MI455X (gfx1250) — matrix operand order repaired and checked by reading, not compiled, not run on hardware
//
#include <hip/hip_runtime.h>
#include <hip/hip_bf16.h>
#include <math.h>

// ---------------- WMMA tile helpers (CDNA5 gfx1250, wave32) ----------------
typedef __attribute__((ext_vector_type(16))) __bf16 v16bf;
typedef __attribute__((ext_vector_type(8)))  float  v8f;

__device__ __forceinline__ v8f v8f_zero() {
    v8f z;
#pragma unroll
    for (int i = 0; i < 8; ++i) z[i] = 0.0f;
    return z;
}

__device__ __forceinline__ v8f wmma_bf16(v16bf a, v16bf b, v8f c) {
    // D = A(16x32) * B(32x16) + C, f32 accumulate
    return __builtin_amdgcn_wmma_f32_16x16x32_bf16(
        /*neg_a=*/false, a, /*neg_b=*/false, b,
        /*c_mod=*/(short)0, c, /*reuse_a=*/false, /*reuse_b=*/false);
}

// A tile 16x32 (MxK), src row-major: elem(m,k) at src[m*ld + k]
// Layout: lanes 0-15 -> K 0..15; lanes 16-31 -> K 16..31 (matches load_bt_tile)
// Per lane 16 contiguous bf16 = 32 contiguous bytes -> 2x b128 loads.
__device__ __forceinline__ v16bf load_a_tile(const __bf16* src, int ld) {
    int lane = threadIdx.x & 31;
    int half = lane >> 4;
    int m    = lane & 15;
    v16bf a;
#pragma unroll
    for (int r = 0; r < 8; ++r) {
        int kb = 2 * r + half * 16;
        const __bf16* p = src + (size_t)m * ld + kb;
        a[2 * r]     = p[0];
        a[2 * r + 1] = p[1];
    }
    return a;
}

// B tile 32x16 (KxN) read from NK (transposed) source: elem(k,n) at src[n*ld + k]
// Per lane 16 contiguous bf16 = 32 contiguous bytes -> 2x b128 loads.
__device__ __forceinline__ v16bf load_bt_tile(const __bf16* src, int ld) {
    int lane = threadIdx.x & 31;
    int half = lane >> 4;
    int n    = lane & 15;
    v16bf b;
#pragma unroll
    for (int r = 0; r < 8; ++r) {
        int k = 2 * r + half * 16;
        const __bf16* p = src + (size_t)n * ld + k;
        b[2 * r]     = p[0];
        b[2 * r + 1] = p[1];
    }
    return b;
}

#define NEG_INF (-3.0e38f)

// ---------------- conversion kernels ----------------
__global__ void cvt_bf16_kernel(const float* __restrict__ in, __bf16* __restrict__ out, int n) {
    int i = blockIdx.x * 256 + threadIdx.x;
    if (i < n) out[i] = (__bf16)in[i];
}

// Batched transpose-convert: in [batch][R][C] f32 -> out [batch][C][R] bf16
__global__ void cvtT_bf16_kernel(const float* __restrict__ in, __bf16* __restrict__ out,
                                 int R, int C, int batch) {
    int i = blockIdx.x * 256 + threadIdx.x;
    int per = R * C;
    if (i < per * batch) {
        int bidx = i / per;
        int rem  = i - bidx * per;
        int r = rem / C, c = rem % C;
        out[(size_t)bidx * per + (size_t)c * R + r] = (__bf16)in[i];
    }
}

// ---------------- LayerNorm (one block per token, D=512) ----------------
__global__ void ln_kernel(const float* __restrict__ x, const float* __restrict__ g,
                          const float* __restrict__ b, float* __restrict__ outf,
                          __bf16* __restrict__ outb) {
    const int D = 512;
    int t = blockIdx.x, tid = threadIdx.x;
    __shared__ float rs[256], rs2[256];
    float s = 0.f, s2 = 0.f;
    for (int d = tid; d < D; d += 256) {
        float v = x[(size_t)t * D + d];
        s += v; s2 += v * v;
    }
    rs[tid] = s; rs2[tid] = s2;
    __syncthreads();
    for (int off = 128; off; off >>= 1) {
        if (tid < off) { rs[tid] += rs[tid + off]; rs2[tid] += rs2[tid + off]; }
        __syncthreads();
    }
    float mu  = rs[0] * (1.0f / D);
    float var = rs2[0] * (1.0f / D) - mu * mu;
    float inv = rsqrtf(var + 1e-5f);
    for (int d = tid; d < D; d += 256) {
        float v = (x[(size_t)t * D + d] - mu) * inv * g[d] + b[d];
        outf[(size_t)t * D + d] = v;
        outb[(size_t)t * D + d] = (__bf16)v;
    }
}

// ---------------- generic WMMA GEMM: one wave -> 16x64 tile ----------------
// N, Kd compile-time so B-tile displacements fold into load immediate offsets.
// K loop unrolled x2 with alternating buffers: no loop-carried register copies.
// Requires Kd % 64 == 0. C[M,N] = A[M,K] * B(NK layout) (+bias) (+resid)
// Optional transposed bf16 output (per-batch [N x trS]) for the V matrix.
template <int N, int Kd>
__global__ void gemm_wmma_kernel(const __bf16* __restrict__ A, const __bf16* __restrict__ Bt,
                                 const float* __restrict__ bias, const float* __restrict__ resid,
                                 float* __restrict__ outf, __bf16* __restrict__ outb,
                                 __bf16* __restrict__ outbT, int trS, int M) {
    constexpr int nbn = N >> 6;
    int nb  = (M >> 4) * nbn;
    int wid = blockIdx.x * (blockDim.x >> 5) + (threadIdx.x >> 5);
    if (wid >= nb) return;
    int mt = wid / nbn, nt = wid % nbn;
    int lane = threadIdx.x & 31, half = lane >> 4, ncol = lane & 15;

    v8f acc[4];
#pragma unroll
    for (int s = 0; s < 4; ++s) acc[s] = v8f_zero();

    const __bf16* Arow  = A + (size_t)mt * 16 * Kd;
    const __bf16* Bbase = Bt + (size_t)(nt * 64) * Kd;

    // prologue: buffer 0 <- K-step 0
    v16bf a0 = load_a_tile(Arow, Kd);
    v16bf b0[4];
#pragma unroll
    for (int s = 0; s < 4; ++s) b0[s] = load_bt_tile(Bbase + (size_t)(s * 16) * Kd, Kd);

    for (int k0 = 0; k0 < Kd; k0 += 64) {
        // buffer 1 <- K-step k0+32 (always in-bounds: Kd % 64 == 0)
        v16bf a1 = load_a_tile(Arow + k0 + 32, Kd);
        v16bf b1[4];
#pragma unroll
        for (int s = 0; s < 4; ++s)
            b1[s] = load_bt_tile(Bbase + (size_t)(s * 16) * Kd + k0 + 32, Kd);
        // consume buffer 0
#pragma unroll
        for (int s = 0; s < 4; ++s) acc[s] = wmma_bf16(a0, b0[s], acc[s]);
        // buffer 0 <- K-step k0+64 (loads write the loop-carried regs directly)
        if (k0 + 64 < Kd) {
            a0 = load_a_tile(Arow + k0 + 64, Kd);
#pragma unroll
            for (int s = 0; s < 4; ++s)
                b0[s] = load_bt_tile(Bbase + (size_t)(s * 16) * Kd + k0 + 64, Kd);
        }
        // consume buffer 1
#pragma unroll
        for (int s = 0; s < 4; ++s) acc[s] = wmma_bf16(a1, b1[s], acc[s]);
    }

#pragma unroll
    for (int s = 0; s < 4; ++s)
#pragma unroll
        for (int r = 0; r < 8; ++r) {
            int m = r + half * 8;
            size_t row = (size_t)mt * 16 + m;
            int col = nt * 64 + s * 16 + ncol;
            float v = acc[s][r];
            if (bias)  v += bias[col];
            if (resid) v += resid[row * N + col];
            if (outf)  outf[row * N + col] = v;
            if (outb)  outb[row * N + col] = (__bf16)v;
            if (outbT) {
                size_t bb = row / trS, ss = row % trS;
                outbT[(bb * N + col) * (size_t)trS + ss] = (__bf16)v;
            }
        }
}

// ---------------- causal attention: one wave per (b, h, 16-query tile) ----------------
// q,k: [BS x D] bf16 row-major;  vT: [B][D][S] bf16 (transposed per batch)
__global__ void attn_kernel(const __bf16* __restrict__ q, const __bf16* __restrict__ k,
                            const __bf16* __restrict__ vT, float* __restrict__ ctx) {
    extern __shared__ char smem[];
    float*  sS = (float*)smem;                           // [16][1024] scores
    __bf16* sP = (__bf16*)(smem + 16 * 1024 * 4);        // [16][1024] probs
    const int S = 1024, D = 512;
    int blk = blockIdx.x;
    int qt = blk & 63, h = (blk >> 6) & 7, b = blk >> 9;
    int lane = threadIdx.x & 31, half = lane >> 4, ncol = lane & 15;
    int qi0 = qt * 16;
    int kvlen = qi0 + 16;
    int nchunk = (kvlen + 31) >> 5;
    int kvpad = nchunk << 5;

    size_t qbase = ((size_t)(b * S + qi0)) * D + h * 64;
    v16bf aq0 = load_a_tile(q + qbase, D);       // K-dim 0..31 of DH=64
    v16bf aq1 = load_a_tile(q + qbase + 32, D);  // K-dim 32..63

    // QK^T: pair loop (both tiles always valid -> no reg copies), odd-tail epilogue.
    // Scores written unconditionally with a select for the causal mask
    // (v_cndmask + ds_store instead of exec-masked stores; no LDS pre-init needed).
    size_t kstride16 = (size_t)16 * D;
    const __bf16* kb_base = k + ((size_t)(b * S)) * D + h * 64;
    int Tq    = qt + 1;
    int Tfull = Tq & ~1;
    v16bf kb0a = load_bt_tile(kb_base, D);
    v16bf kb0b = load_bt_tile(kb_base + 32, D);
    for (int jt = 0; jt < Tfull; jt += 2) {
        v16bf kb1a = load_bt_tile(kb_base + (size_t)(jt + 1) * kstride16, D);
        v16bf kb1b = load_bt_tile(kb_base + (size_t)(jt + 1) * kstride16 + 32, D);
        v8f c = v8f_zero();
        c = wmma_bf16(aq0, kb0a, c);
        c = wmma_bf16(aq1, kb0b, c);
#pragma unroll
        for (int r = 0; r < 8; ++r) {
            int m = r + half * 8;
            int j = jt * 16 + ncol;
            sS[m * 1024 + j] = (j <= qi0 + m) ? c[r] * 0.125f : NEG_INF;
        }
        if (jt + 2 < Tq) {
            kb0a = load_bt_tile(kb_base + (size_t)(jt + 2) * kstride16, D);
            kb0b = load_bt_tile(kb_base + (size_t)(jt + 2) * kstride16 + 32, D);
        }
        v8f c2 = v8f_zero();
        c2 = wmma_bf16(aq0, kb1a, c2);
        c2 = wmma_bf16(aq1, kb1b, c2);
#pragma unroll
        for (int r = 0; r < 8; ++r) {
            int m = r + half * 8;
            int j = (jt + 1) * 16 + ncol;
            sS[m * 1024 + j] = (j <= qi0 + m) ? c2[r] * 0.125f : NEG_INF;
        }
    }
    if (Tq & 1) {  // last tile (index Tq-1) sits in buffer 0
        v8f c = v8f_zero();
        c = wmma_bf16(aq0, kb0a, c);
        c = wmma_bf16(aq1, kb0b, c);
#pragma unroll
        for (int r = 0; r < 8; ++r) {
            int m = r + half * 8;
            int j = (Tq - 1) * 16 + ncol;
            sS[m * 1024 + j] = (j <= qi0 + m) ? c[r] * 0.125f : NEG_INF;
        }
    }
    __syncthreads();

    // softmax: 2 lanes per row
    int row = lane >> 1, sub = lane & 1;
    float mx = NEG_INF;
    for (int j = sub; j < kvlen; j += 2) mx = fmaxf(mx, sS[row * 1024 + j]);
    mx = fmaxf(mx, __shfl_xor(mx, 1));
    float sum = 0.f;
    for (int j = sub; j < kvlen; j += 2) {
        float e = __expf(sS[row * 1024 + j] - mx);
        sum += e;
        sS[row * 1024 + j] = e;
    }
    sum += __shfl_xor(sum, 1);
    float inv = 1.0f / sum;
    for (int j = sub; j < kvpad; j += 2)
        sP[row * 1024 + j] = (j < kvlen) ? (__bf16)(sS[row * 1024 + j] * inv) : (__bf16)0.0f;
    __syncthreads();

    // O = P * V: pair loop + odd-tail epilogue (no reg copies)
    v8f acc[4];
#pragma unroll
    for (int s = 0; s < 4; ++s) acc[s] = v8f_zero();
    const __bf16* vbase = vT + ((size_t)b * D + h * 64) * S;
    int nfull = nchunk & ~1;
    v16bf p0 = load_a_tile(sP, 1024);
    v16bf v0[4];
#pragma unroll
    for (int s = 0; s < 4; ++s) v0[s] = load_bt_tile(vbase + (size_t)(s * 16) * S, S);
    for (int ch = 0; ch < nfull; ch += 2) {
        v16bf p1 = load_a_tile(sP + (ch + 1) * 32, 1024);
        v16bf v1[4];
#pragma unroll
        for (int s = 0; s < 4; ++s)
            v1[s] = load_bt_tile(vbase + (size_t)(s * 16) * S + (ch + 1) * 32, S);
#pragma unroll
        for (int s = 0; s < 4; ++s) acc[s] = wmma_bf16(p0, v0[s], acc[s]);
        if (ch + 2 < nchunk) {
            p0 = load_a_tile(sP + (ch + 2) * 32, 1024);
#pragma unroll
            for (int s = 0; s < 4; ++s)
                v0[s] = load_bt_tile(vbase + (size_t)(s * 16) * S + (ch + 2) * 32, S);
        }
#pragma unroll
        for (int s = 0; s < 4; ++s) acc[s] = wmma_bf16(p1, v1[s], acc[s]);
    }
    if (nchunk & 1) {  // last chunk sits in buffer 0
#pragma unroll
        for (int s = 0; s < 4; ++s) acc[s] = wmma_bf16(p0, v0[s], acc[s]);
    }
#pragma unroll
    for (int s = 0; s < 4; ++s)
#pragma unroll
        for (int r = 0; r < 8; ++r) {
            int m = r + half * 8;
            ctx[((size_t)(b * S + qi0 + m)) * D + h * 64 + s * 16 + ncol] = acc[s][r];
        }
}

// ---------------- router: gate, score mix, top-16, neuron residual, coords ----------------
__global__ void route_topk_kernel(const float* __restrict__ x, const float* __restrict__ normed,
                                  const float* __restrict__ ctx, const float* __restrict__ ts,
                                  const float* __restrict__ cs, const float* __restrict__ Wg,
                                  const float* __restrict__ bg, const float* __restrict__ emb,
                                  const float* __restrict__ coords, float* __restrict__ xA,
                                  float* __restrict__ tc, int* __restrict__ outIdx) {
    const int D = 512, NN = 512, K = 16, NB = 8;
    int t = blockIdx.x, tid = threadIdx.x;
    __shared__ float sc[512];
    __shared__ float r0[256], r1[256];
    __shared__ float kval[16];
    __shared__ int   kidx[16];
    __shared__ float wsh[16];

    // gate logits over concat(normed, ctx) @ Wg[1024x2]
    float a0 = 0.f, a1 = 0.f;
    for (int i = tid; i < 2 * D; i += 256) {
        float vi = (i < D) ? normed[(size_t)t * D + i] : ctx[(size_t)t * D + (i - D)];
        a0 += vi * Wg[i * 2 + 0];
        a1 += vi * Wg[i * 2 + 1];
    }
    r0[tid] = a0; r1[tid] = a1;
    __syncthreads();
    for (int off = 128; off; off >>= 1) {
        if (tid < off) { r0[tid] += r0[tid + off]; r1[tid] += r1[tid + off]; }
        __syncthreads();
    }
    float l0 = r0[0] + bg[0], l1 = r1[0] + bg[1];
    float mg = fmaxf(l0, l1);
    float e0 = __expf(l0 - mg), e1 = __expf(l1 - mg);
    float g0 = e0 / (e0 + e1), g1 = e1 / (e0 + e1);

    for (int n = tid; n < NN; n += 256)
        sc[n] = g0 * ts[(size_t)t * NN + n] + g1 * cs[(size_t)t * NN + n];
    __syncthreads();

    if (tid < 32) {
        for (int kk = 0; kk < K; ++kk) {
            float best = NEG_INF;
            int   bi   = 0x7fffffff;
            for (int n = tid; n < NN; n += 32) {
                float vv = sc[n];
                if (vv > best || (vv == best && n < bi)) { best = vv; bi = n; }
            }
            for (int off = 16; off; off >>= 1) {
                float ov = __shfl_xor(best, off);
                int   oi = __shfl_xor(bi, off);
                if (ov > best || (ov == best && oi < bi)) { best = ov; bi = oi; }
            }
            if (tid == 0) { kval[kk] = best; kidx[kk] = bi; sc[bi] = NEG_INF; }
        }
        if (tid < 16) {
            float vv = kval[tid];
            float m2 = vv;
            for (int off = 8; off; off >>= 1) m2 = fmaxf(m2, __shfl_xor(m2, off, 16));
            float e = __expf(vv - m2);
            float s2 = e;
            for (int off = 8; off; off >>= 1) s2 += __shfl_xor(s2, off, 16);
            wsh[tid] = e / s2;
        }
    }
    __syncthreads();

    for (int d = tid; d < D; d += 256) {
        float acc = x[(size_t)t * D + d];
        for (int kk = 0; kk < K; ++kk) acc += wsh[kk] * emb[(size_t)kidx[kk] * D + d];
        xA[(size_t)t * D + d] = acc;
    }
    if (tid < NB) {
        float c = 0.f;
        for (int kk = 0; kk < K; ++kk) c += wsh[kk] * coords[(size_t)kidx[kk] * NB + tid];
        tc[(size_t)t * NB + tid] = c;
    }
    if (tid < K) outIdx[(size_t)t * K + tid] = kidx[tid];
}

// ---------------- basis-mixed low-rank FFN: stage 1 (D->R, weighted over bases) ----------------
// X: [BS x 512] bf16; AbT: [8][64 x 512] bf16 (NK); tcv: [BS x 8]; Hout: [BS x 64] bf16
__global__ void mix_gemm_A_kernel(const __bf16* __restrict__ X, const __bf16* __restrict__ AbT,
                                  const float* __restrict__ tcv, __bf16* __restrict__ Hout) {
    const int Kd = 512, R = 64, NB = 8;
    int wid = blockIdx.x * (blockDim.x >> 5) + (threadIdx.x >> 5);
    if (wid >= 128) return;
    int mt = wid;
    int lane = threadIdx.x & 31, half = lane >> 4, ncol = lane & 15;
    v8f acc[4];
#pragma unroll
    for (int s = 0; s < 4; ++s) acc[s] = v8f_zero();

    const __bf16* Xrow = X + (size_t)(mt * 16) * Kd;

    for (int n = 0; n < NB; ++n) {
        float cv[8];
#pragma unroll
        for (int r = 0; r < 8; ++r) cv[r] = tcv[(size_t)(mt * 16 + r + half * 8) * NB + n];
        const __bf16* Bb = AbT + (size_t)n * R * Kd;
        v8f tacc[4];
#pragma unroll
        for (int s = 0; s < 4; ++s) tacc[s] = v8f_zero();

        v16bf a0 = load_a_tile(Xrow, Kd);
        v16bf b0[4];
#pragma unroll
        for (int s = 0; s < 4; ++s) b0[s] = load_bt_tile(Bb + (size_t)(s * 16) * Kd, Kd);
        for (int k0 = 0; k0 < Kd; k0 += 64) {
            v16bf a1 = load_a_tile(Xrow + k0 + 32, Kd);
            v16bf b1[4];
#pragma unroll
            for (int s = 0; s < 4; ++s)
                b1[s] = load_bt_tile(Bb + (size_t)(s * 16) * Kd + k0 + 32, Kd);
#pragma unroll
            for (int s = 0; s < 4; ++s) tacc[s] = wmma_bf16(a0, b0[s], tacc[s]);
            if (k0 + 64 < Kd) {
                a0 = load_a_tile(Xrow + k0 + 64, Kd);
#pragma unroll
                for (int s = 0; s < 4; ++s)
                    b0[s] = load_bt_tile(Bb + (size_t)(s * 16) * Kd + k0 + 64, Kd);
            }
#pragma unroll
            for (int s = 0; s < 4; ++s) tacc[s] = wmma_bf16(a1, b1[s], tacc[s]);
        }
#pragma unroll
        for (int s = 0; s < 4; ++s)
#pragma unroll
            for (int r = 0; r < 8; ++r) acc[s][r] += cv[r] * tacc[s][r];
    }
#pragma unroll
    for (int s = 0; s < 4; ++s)
#pragma unroll
        for (int r = 0; r < 8; ++r) {
            int m = r + half * 8;
            Hout[(size_t)(mt * 16 + m) * R + s * 16 + ncol] = (__bf16)acc[s][r];
        }
}

// ---------------- basis-mixed low-rank FFN: stage 2 (R->D_FF, weighted, + exact GeLU) ----------------
// Hin: [BS x 64] bf16; BbT: [8][2048 x 64] bf16 (NK); HFout: [BS x 2048] bf16
__global__ void mix_gemm_B_kernel(const __bf16* __restrict__ Hin, const __bf16* __restrict__ BbT,
                                  const float* __restrict__ tcv, __bf16* __restrict__ HFout) {
    const int Kd = 64, DFF = 2048, NB = 8;
    int wid = blockIdx.x * (blockDim.x >> 5) + (threadIdx.x >> 5);
    if (wid >= 128 * 32) return;
    int mt = wid >> 5, ft = wid & 31;
    int lane = threadIdx.x & 31, half = lane >> 4, ncol = lane & 15;
    v8f acc[4];
#pragma unroll
    for (int s = 0; s < 4; ++s) acc[s] = v8f_zero();

    // A tiles reused across all 8 bases: K=64 -> 2 tiles, load once
    v16bf a0 = load_a_tile(Hin + (size_t)(mt * 16) * Kd, Kd);
    v16bf a1 = load_a_tile(Hin + (size_t)(mt * 16) * Kd + 32, Kd);

    for (int n = 0; n < NB; ++n) {
        float cv[8];
#pragma unroll
        for (int r = 0; r < 8; ++r) cv[r] = tcv[(size_t)(mt * 16 + r + half * 8) * NB + n];
        const __bf16* Bb = BbT + (size_t)n * DFF * Kd + (size_t)(ft * 64) * Kd;
        v8f tacc[4];
#pragma unroll
        for (int s = 0; s < 4; ++s) tacc[s] = v8f_zero();
#pragma unroll
        for (int s = 0; s < 4; ++s) {
            v16bf b0 = load_bt_tile(Bb + (size_t)(s * 16) * Kd, Kd);
            v16bf b1 = load_bt_tile(Bb + (size_t)(s * 16) * Kd + 32, Kd);
            tacc[s] = wmma_bf16(a0, b0, tacc[s]);
            tacc[s] = wmma_bf16(a1, b1, tacc[s]);
        }
#pragma unroll
        for (int s = 0; s < 4; ++s)
#pragma unroll
            for (int r = 0; r < 8; ++r) acc[s][r] += cv[r] * tacc[s][r];
    }
#pragma unroll
    for (int s = 0; s < 4; ++s)
#pragma unroll
        for (int r = 0; r < 8; ++r) {
            int m = r + half * 8;
            float v = acc[s][r];
            float g = 0.5f * v * (1.0f + erff(v * 0.70710678118654752f));  // exact GeLU
            HFout[(size_t)(mt * 16 + m) * DFF + ft * 64 + s * 16 + ncol] = (__bf16)g;
        }
}

// ---------------- host launch ----------------
extern "C" void kernel_launch(void* const* d_in, const int* in_sizes, int n_in,
                              void* d_out, int out_size, void* d_ws, size_t ws_size,
                              hipStream_t stream) {
    const int B = 2, S = 1024, D = 512, BS = B * S, NN = 512, K = 16, NB = 8, R = 64, DFF = 2048;

    const float* x_in   = (const float*)d_in[0];
    const float* emb    = (const float*)d_in[1];
    const float* Wq     = (const float*)d_in[2];
    const float* bq     = (const float*)d_in[3];
    const float* Wk     = (const float*)d_in[4];
    const float* bk     = (const float*)d_in[5];
    const float* Wv     = (const float*)d_in[6];
    const float* bv     = (const float*)d_in[7];
    const float* Wg     = (const float*)d_in[8];
    const float* bg     = (const float*)d_in[9];
    const float* bA     = (const float*)d_in[10];
    const float* bB     = (const float*)d_in[11];
    const float* coords = (const float*)d_in[12];
    const float* Wd     = (const float*)d_in[13];
    const float* bd     = (const float*)d_in[14];
    const float* ln1g   = (const float*)d_in[15];
    const float* ln1b   = (const float*)d_in[16];
    const float* ln2g   = (const float*)d_in[17];
    const float* ln2b   = (const float*)d_in[18];
    // d_in[19] = causal mask (implicit in kernel)

    float* outX   = (float*)d_out;
    int*   outIdx = (int*)(outX + (size_t)BS * D);

    // workspace layout
    char* w = (char*)d_ws;
    size_t off = 0;
    auto alloc = [&](size_t bytes) {
        size_t o = off;
        off += (bytes + 255) & ~(size_t)255;
        return (void*)(w + o);
    };
    float*  normedF = (float*)alloc((size_t)BS * D * 4);
    __bf16* normedB = (__bf16*)alloc((size_t)BS * D * 2);
    __bf16* wqT = (__bf16*)alloc((size_t)D * D * 2);       // [N x K]
    __bf16* wkT = (__bf16*)alloc((size_t)D * D * 2);
    __bf16* wvT = (__bf16*)alloc((size_t)D * D * 2);
    __bf16* wdT = (__bf16*)alloc((size_t)D * DFF * 2);     // [512 x 2048]
    __bf16* baT = (__bf16*)alloc((size_t)NB * R * D * 2);  // [8][64 x 512]
    __bf16* bbT = (__bf16*)alloc((size_t)NB * DFF * R * 2);// [8][2048 x 64]
    __bf16* embB = (__bf16*)alloc((size_t)NN * D * 2);     // natural layout = NK
    __bf16* qB = (__bf16*)alloc((size_t)BS * D * 2);
    __bf16* kB = (__bf16*)alloc((size_t)BS * D * 2);
    __bf16* vT = (__bf16*)alloc((size_t)BS * D * 2);       // [B][D][S]
    float*  ctxF = (float*)alloc((size_t)BS * D * 4);
    __bf16* ctxB = (__bf16*)alloc((size_t)BS * D * 2);
    float*  tsF = (float*)alloc((size_t)BS * NN * 4);
    float*  csF = (float*)alloc((size_t)BS * NN * 4);
    float*  xA  = (float*)alloc((size_t)BS * D * 4);
    float*  tcF = (float*)alloc((size_t)BS * NB * 4);
    float*  n2F = (float*)alloc((size_t)BS * D * 4);
    __bf16* n2B = (__bf16*)alloc((size_t)BS * D * 2);
    __bf16* hB  = (__bf16*)alloc((size_t)BS * R * 2);
    __bf16* hfB = (__bf16*)alloc((size_t)BS * DFF * 2);
    (void)ws_size; (void)n_in; (void)in_sizes; (void)out_size;

    auto cvt = [&](const float* src, __bf16* dst, int n) {
        cvt_bf16_kernel<<<(n + 255) / 256, 256, 0, stream>>>(src, dst, n);
    };
    auto cvtT = [&](const float* src, __bf16* dst, int Rr, int Cc, int batch) {
        int n = Rr * Cc * batch;
        cvtT_bf16_kernel<<<(n + 255) / 256, 256, 0, stream>>>(src, dst, Rr, Cc, batch);
    };
    // weight conversions (all B operands stored NK for contiguous b128 tile loads)
    cvtT(Wq, wqT, D, D, 1);
    cvtT(Wk, wkT, D, D, 1);
    cvtT(Wv, wvT, D, D, 1);
    cvtT(Wd, wdT, DFF, D, 1);
    cvtT(bA, baT, D, R, NB);
    cvtT(bB, bbT, R, DFF, NB);
    cvt(emb, embB, NN * D);

    // LN1
    ln_kernel<<<BS, 256, 0, stream>>>(x_in, ln1g, ln1b, normedF, normedB);

    // QKV projections (Q,K row layout; V written transposed per batch for PV stage)
    int gemmBlk = ((BS / 16) * (D / 64) + 7) / 8;
    gemm_wmma_kernel<512, 512><<<gemmBlk, 256, 0, stream>>>(normedB, wqT, bq, nullptr, nullptr,
                                                            qB, nullptr, 0, BS);
    gemm_wmma_kernel<512, 512><<<gemmBlk, 256, 0, stream>>>(normedB, wkT, bk, nullptr, nullptr,
                                                            kB, nullptr, 0, BS);
    gemm_wmma_kernel<512, 512><<<gemmBlk, 256, 0, stream>>>(normedB, wvT, bv, nullptr, nullptr,
                                                            nullptr, vT, S, BS);

    // attention (1 wave per (b,h,qtile)); 96KB dynamic LDS
    attn_kernel<<<B * 8 * (S / 16), 32, 96 * 1024, stream>>>(qB, kB, vT, ctxF);
    cvt(ctxF, ctxB, BS * D);

    // neuron scores (emb natural [N x D] layout is exactly NK)
    gemm_wmma_kernel<512, 512><<<gemmBlk, 256, 0, stream>>>(normedB, embB, nullptr, nullptr,
                                                            tsF, nullptr, nullptr, 0, BS);
    gemm_wmma_kernel<512, 512><<<gemmBlk, 256, 0, stream>>>(ctxB, embB, nullptr, nullptr,
                                                            csF, nullptr, nullptr, 0, BS);

    // routing + top-k + neuron residual + token coords + idx output
    route_topk_kernel<<<BS, 256, 0, stream>>>(x_in, normedF, ctxF, tsF, csF, Wg, bg, emb,
                                              coords, xA, tcF, outIdx);

    // LN2
    ln_kernel<<<BS, 256, 0, stream>>>(xA, ln2g, ln2b, n2F, n2B);

    // basis-mixed FFN
    mix_gemm_A_kernel<<<(128 + 7) / 8, 256, 0, stream>>>(n2B, baT, tcF, hB);
    mix_gemm_B_kernel<<<(128 * 32 + 7) / 8, 256, 0, stream>>>(hB, bbT, tcF, hfB);

    // final projection + bias + residual -> d_out x part
    gemm_wmma_kernel<512, 2048><<<gemmBlk, 256, 0, stream>>>(hfB, wdT, bd, xA, outX, nullptr,
                                                             nullptr, 0, BS);
}
